// Hybrid_Conv2d_48524540510660
// MI455X (gfx1250) — compile-verified, matrix operand order checked by reading, not run on hardware
//
#include <hip/hip_runtime.h>

// ---------------------------------------------------------------------------
// Hybrid conv2d (per-sample kernel W0 + cov[b]*W1), implicit GEMM on CDNA5.
// bf16 WMMA path (v_wmma_f32_16x16x32_bf16), f32 accumulation over K=576.
//
// K-order: k = (kh*3+kw)*CIN + ci  ->  each 32-K chunk has a uniform kernel
// tap (rc = kt>>1) and ci = (kt&1)*32 + kl. Within a lane-row the 16 fragment
// elements differ only by ci, i.e. a compile-time kl*HW offset -> im2col is
// 16 immediate-offset b32 loads off one base pointer, one bounds test.
//
// Weights are pre-combined (W0 + cov[b]*W1 -> bf16) in WMMA fragment order
// into d_ws by a small prep kernel; the main loop reads A fragments straight
// from global (L2-resident, 73.7 KB/batch reused by 64 blocks).
//
// fp32->bf16 uses the hardware packed conversion (v_cvt_pk_bf16_f32, RNE)
// via a 2-element __bf16 ext-vector.
// ---------------------------------------------------------------------------

typedef __attribute__((ext_vector_type(16))) __bf16 v16bf;
typedef __attribute__((ext_vector_type(2)))  __bf16 v2bf;
typedef __attribute__((ext_vector_type(8)))  float  v8f;

#define BATCH 32
#define CIN   64
#define COUT  64
#define HH    128
#define WW    128
#define HW    (HH * WW)        // 16384
#define KTOT  (CIN * 9)        // 576
#define KSTEP 32               // K per WMMA
#define NKT   (KTOT / KSTEP)   // 18
#define NTILE 256              // pixels per workgroup (2 rows of 128)

// packed fp32x2 -> bf16x2 (hardware cvt, round-to-nearest-even)
__device__ __forceinline__ unsigned pack_bf16(float lo, float hi) {
  v2bf t;
  t[0] = (__bf16)lo;
  t[1] = (__bf16)hi;
  return __builtin_bit_cast(unsigned, t);
}

// Fragment-order mapping for 16-bit 16x32 tiles (ISA 7.12.2):
//   element e (0..15), lane: K = ((lane>>4)<<3) + (e&7) + ((e&8)<<1)

// ---------------------------------------------------------------------------
// Prep kernel: combined weights -> bf16, fragment order, into d_ws.
// Layout: wfrag[((b*NKT + kt)*4 + mtile)*512 + lane*16 + e]
// One block per (b, kt): 4 mtiles * 512 = 2048 elems, 256 thr * 8.
// ---------------------------------------------------------------------------
__global__ __launch_bounds__(256)
void prep_weights(const float* __restrict__ W0, const float* __restrict__ W1,
                  const float* __restrict__ cov,
                  unsigned short* __restrict__ wfrag) {
  const int b      = blockIdx.x / NKT;
  const int kt     = blockIdx.x % NKT;
  const float covb = cov[b];
  const int rc     = kt >> 1;          // kernel tap (kh*3+kw)
  const int cibase = (kt & 1) * 32;
  const int tid    = threadIdx.x;

  unsigned tmp[4];
#pragma unroll
  for (int i2 = 0; i2 < 4; ++i2) {
    // pair (e, e+1): consecutive kl -> ci and ci+1 (widx stride 9)
    const int flat = tid * 8 + i2 * 2;
    const int e    = flat & 15;
    const int lane = (flat >> 4) & 31;
    const int mt   = flat >> 9;
    const int kl   = ((lane >> 4) << 3) + (e & 7) + ((e & 8) << 1);
    const int ci   = cibase + kl;
    const int co   = mt * 16 + (lane & 15);
    const int widx = co * KTOT + ci * 9 + rc;   // OIHW flat
    const float w0 = W0[widx]     + covb * W1[widx];
    const float w1 = W0[widx + 9] + covb * W1[widx + 9];
    tmp[i2] = pack_bf16(w0, w1);
  }
  uint4* dst = reinterpret_cast<uint4*>(
      &wfrag[(size_t)blockIdx.x * 2048 + tid * 8]);
  *dst = *reinterpret_cast<const uint4*>(tmp);
}

// ---------------------------------------------------------------------------
// im2col producer: one 32-K chunk of B[32 x 256] in fragment order.
// Each thread services its own wave lane for tiles nt = wave and wave+8, so
// at each load instruction 16 consecutive lanes hit 16 consecutive pixels
// (two full 64B segments). Per lane-row: one bounds test, one base pointer,
// 16 immediate-offset loads (kl*HW*4 <= 6 MB < imm24).
// ---------------------------------------------------------------------------
__device__ __forceinline__ void fill_B(unsigned short* __restrict__ ldsB,
                                       const float* __restrict__ xb,
                                       int pix0, int kt, int tid) {
  const int rc     = kt >> 1;          // uniform within chunk (SALU)
  const int r      = rc / 3;
  const int c      = rc - r * 3;
  const int cibase = (kt & 1) * 32;
  const int lane   = tid & 31;
  const int wv     = tid >> 5;
  const int lanehi = lane >> 4;

#pragma unroll
  for (int j = 0; j < 2; ++j) {
    const int nt  = wv + j * 8;
    const int n   = nt * 16 + (lane & 15);
    const int p   = pix0 + n;
    const int h   = (p >> 7) + r - 1;
    const int w   = (p & 127) + c - 1;
    const bool ok = ((unsigned)h < (unsigned)HH) & ((unsigned)w < (unsigned)WW);
    const float* bp = xb + (size_t)(cibase + lanehi * 8) * HW + (h * WW + w);

    unsigned pk[8];
#pragma unroll
    for (int e2 = 0; e2 < 8; ++e2) {
      // e=2*e2 -> kl-lanehi*8 = {0,2,4,6,16,18,20,22}; odd e = +1
      const int kl0 = (e2 < 4) ? (2 * e2) : (2 * e2 + 8);
      const float v0 = ok ? bp[(size_t)kl0 * HW] : 0.0f;
      const float v1 = ok ? bp[(size_t)(kl0 + 1) * HW] : 0.0f;
      pk[e2] = pack_bf16(v0, v1);
    }
    uint4* dst = reinterpret_cast<uint4*>(&ldsB[nt * 512 + lane * 16]);
    dst[0] = reinterpret_cast<const uint4*>(pk)[0];
    dst[1] = reinterpret_cast<const uint4*>(pk)[1];
  }
}

// ---------------------------------------------------------------------------
// Main kernel
// ---------------------------------------------------------------------------
__global__ __launch_bounds__(256)
void hybrid_conv2d_wmma(const float* __restrict__ x,
                        const unsigned short* __restrict__ wfrag,
                        float* __restrict__ out) {
  // Double-buffered B staging only: 2 * 16 tiles * 512 bf16 = 32 KB
  __shared__ __attribute__((aligned(64))) unsigned short ldsB[2][16 * 512];

  const int tid  = threadIdx.x;
  const int lane = tid & 31;
  const int wave = tid >> 5;

  const int b    = blockIdx.x / (HW / NTILE);
  const int tile = blockIdx.x % (HW / NTILE);
  const int pix0 = tile * NTILE;

  const float* xb = x + (size_t)b * CIN * HW;

  const int m_strip = wave & 3;   // which 16 output channels
  const int n_half  = wave >> 2;  // which 128-pixel half of the tile

  v8f acc[8];
#pragma unroll
  for (int i = 0; i < 8; ++i) acc[i] = v8f{};

  fill_B(ldsB[0], xb, pix0, 0, tid);
  __syncthreads();

  const int b_off = n_half * 8 * 512 + lane * 16;
  const unsigned short* wA =
      wfrag + ((size_t)b * NKT * 4 + m_strip) * 512 + lane * 16;

  for (int kt = 0; kt < NKT; ++kt) {
    const int cur = kt & 1;

    // (a) all fragment loads first: bfrags from LDS (in-order ds, ahead of the
    //     producer's ds stores), afrag straight from global (L2-resident)
    const v16bf afrag =
        *reinterpret_cast<const v16bf*>(wA + (size_t)kt * 2048);
    v16bf bfrag[8];
#pragma unroll
    for (int nt = 0; nt < 8; ++nt)
      bfrag[nt] = *reinterpret_cast<const v16bf*>(&ldsB[cur][b_off + nt * 512]);

    // (b) producer for next chunk; global latency hidden under WMMA chain
    if (kt + 1 < NKT) fill_B(ldsB[cur ^ 1], xb, pix0, kt + 1, tid);

    // (c) 8 WMMAs on held fragments
#pragma unroll
    for (int nt = 0; nt < 8; ++nt) {
      acc[nt] = __builtin_amdgcn_wmma_f32_16x16x32_bf16(
          /*neg_a=*/false, afrag, /*neg_b=*/false, bfrag[nt],
          /*c_mod=*/(short)0, acc[nt], /*reuse_a=*/false, /*reuse_b=*/false);
    }
    __syncthreads();
  }

  // Epilogue: C/D layout -> N = lane&15, M = v + 8*(lane>=16). Per (nt,v) the
  // 32 lanes cover two contiguous 64B row segments. Non-temporal: output is
  // write-once; keep L2 for the reused x.
  float* outb = out + (size_t)b * COUT * HW;
  const int nlo = lane & 15;
  const int mhi = (lane >> 4) << 3;
#pragma unroll
  for (int nt = 0; nt < 8; ++nt) {
    const int p = pix0 + n_half * 128 + nt * 16 + nlo;
#pragma unroll
    for (int v = 0; v < 8; ++v) {
      const int co = m_strip * 16 + mhi + v;
      __builtin_nontemporal_store(acc[nt][v], &outb[(size_t)co * HW + p]);
    }
  }
}

extern "C" void kernel_launch(void* const* d_in, const int* in_sizes, int n_in,
                              void* d_out, int out_size, void* d_ws, size_t ws_size,
                              hipStream_t stream) {
  const float* x  = (const float*)d_in[0];
  const float* cv = (const float*)d_in[1];
  const float* W0 = (const float*)d_in[2];
  const float* W1 = (const float*)d_in[3];
  float* out      = (float*)d_out;
  unsigned short* wfrag = (unsigned short*)d_ws;  // 32*18*2048*2 B = 2.36 MB

  // 1) combine + transpose weights into fragment order (bf16)
  prep_weights<<<dim3(BATCH * NKT), dim3(256), 0, stream>>>(W0, W1, cv, wfrag);
  // 2) implicit-GEMM conv
  hybrid_conv2d_wmma<<<dim3(BATCH * (HW / NTILE)), dim3(256), 0, stream>>>(
      x, wfrag, out);
}
